// GLCA_3015067042108
// MI455X (gfx1250) — matrix operand order repaired and checked by reading, not compiled, not run on hardware
//
#include <hip/hip_runtime.h>

// ---------------------------------------------------------------------------
// GLCA (global/local channel attention) for MI455X / gfx1250.
//
// Roofline: ~0.8 GB of HBM traffic (2x read of x + 1x write) => ~34 us at
// 23.3 TB/s. The per-batch 256x256 @ 256x4096 GEMM (17.2 GFLOP total) runs
// on v_wmma_f32_16x16x32_bf16; the 4 MiB per-batch XL working set is
// L2-resident (192 MB L2) so operand re-reads don't touch HBM.
// ---------------------------------------------------------------------------

typedef __attribute__((ext_vector_type(16))) __bf16 v16bf;
typedef __attribute__((ext_vector_type(8)))  __bf16 v8bf;
typedef __attribute__((ext_vector_type(8)))  float  v8f;

#define N_B   32
#define C_ALL 512
#define CH    256
#define HW    4096

// Branchless RNE f32 -> bf16 (inputs are finite: no NaN guard needed).
static __device__ __forceinline__ unsigned short f32_to_bf16_bits(float f) {
    unsigned int u = __float_as_uint(f);
    u += 0x7fffu + ((u >> 16) & 1u);
    return (unsigned short)(u >> 16);
}
static __device__ __forceinline__ unsigned int pack_bf16x2(float lo, float hi) {
    return (unsigned int)f32_to_bf16_bits(lo) |
           ((unsigned int)f32_to_bf16_bits(hi) << 16);
}

// ---------------------------------------------------------------------------
// Kernel 1: per-channel mean over HxW.  One block per (n,c).
// ---------------------------------------------------------------------------
__global__ void __launch_bounds__(256) k_mean(const float* __restrict__ x,
                                              float* __restrict__ g) {
    __shared__ float red[256];
    const size_t base = (size_t)blockIdx.x * HW;
    const int t = threadIdx.x;
    const float4* xv = (const float4*)(x + base);
    float s = 0.f;
#pragma unroll
    for (int j = 0; j < 4; ++j) {
        float4 v = xv[t + 256 * j];
        s += v.x + v.y + v.z + v.w;
    }
    red[t] = s;
    __syncthreads();
    for (int off = 128; off > 0; off >>= 1) {
        if (t < off) red[t] += red[t + off];
        __syncthreads();
    }
    if (t == 0) g[blockIdx.x] = red[0] * (1.0f / 4096.0f);
}

// ---------------------------------------------------------------------------
// Kernel 2: conv1d(K=5)+sigmoid on channel means, softmax(outer(q,q)), and
// write qk (symmetric => A-matrix == qk) pre-swizzled into the WMMA bf16
// 16x32 A-fragment lane layout:
//   halves[idx], idx = ((mt*8 + kt)*32 + lane)*16 + h
//   M = mt*16 + (lane&15), K = kt*32 + h + 16*(lane>>4)
// Also writes the right-half scale (al + 2).  One block per batch n.
// ---------------------------------------------------------------------------
__global__ void __launch_bounds__(256) k_attn_prep(
        const float* __restrict__ g,
        const float* __restrict__ w_q,  const float* __restrict__ b_q,
        const float* __restrict__ w_l,  const float* __restrict__ b_l,
        float* __restrict__ scale_r,
        unsigned short* __restrict__ qkA) {
    __shared__ float sq[CH];
    __shared__ float red[256];
    __shared__ float s_m2, s_invS;
    const int n = blockIdx.x;
    const int c = threadIdx.x;
    const float* gl = g + n * C_ALL;   // means of left-half channels
    const float* gr = gl + CH;         // means of right-half channels

    float accq = b_q[0], accl = b_l[0];
#pragma unroll
    for (int k = 0; k < 5; ++k) {      // cross-correlation, 'same' zero pad
        int j = c + k - 2;
        if (j >= 0 && j < CH) {
            accq += gl[j] * w_q[k];
            accl += gr[j] * w_l[k];
        }
    }
    float q  = 1.f / (1.f + expf(-accq));
    float al = 1.f / (1.f + expf(-accl));
    sq[c] = q;
    scale_r[n * CH + c] = al + 2.0f;

    // ---- max(q) -> m2 = max(q)^2  (q in (0,1), all positive) ----
    red[c] = q;
    __syncthreads();
    for (int off = 128; off > 0; off >>= 1) {
        if (c < off) red[c] = fmaxf(red[c], red[c + off]);
        __syncthreads();
    }
    if (c == 0) s_m2 = red[0] * red[0];
    __syncthreads();
    const float m2 = s_m2;

    // ---- S = sum_ij exp(q_i q_j - m2) ----
    float part = 0.f;
    const float qc = sq[c];
    for (int j = 0; j < CH; ++j) part += expf(qc * sq[j] - m2);
    red[c] = part;
    __syncthreads();
    for (int off = 128; off > 0; off >>= 1) {
        if (c < off) red[c] += red[c + off];
        __syncthreads();
    }
    if (c == 0) s_invS = 1.0f / red[0];
    __syncthreads();
    const float invS = s_invS;

    // ---- swizzled bf16 A-fragments ----
    unsigned short* dst = qkA + (size_t)n * (CH * CH);
    for (int i = 0; i < 256; ++i) {
        int idx  = c + 256 * i;        // coalesced across threads
        int h    = idx & 15;
        int lane = (idx >> 4) & 31;
        int kt   = (idx >> 9) & 7;
        int mt   = (idx >> 12) & 15;
        int M = mt * 16 + (lane & 15);
        int K = kt * 32 + h + ((lane >> 4) << 4);
        float v = expf(sq[M] * sq[K] - m2) * invS;
        dst[idx] = f32_to_bf16_bits(v);
    }
}

// ---------------------------------------------------------------------------
// Kernel 3: att = qk @ XL via v_wmma_f32_16x16x32_bf16, fused epilogue
// out_left = xl * (1 + att).
// Block tile: M=64 channels x N=128 pixels, 8 waves (4 in M, 2 in N),
// each wave owns a 16x64 output strip (4 accumulators).
// B operand (xl) is double-buffered in LDS as bf16, transposed to
// [pixel][k] with a 40-half padded stride: each lane's fragment is one
// contiguous, 16B-aligned 32-byte read.  Staging: each thread pulls two
// float4 from adjacent K-rows and emits 4 packed ds_store_b32.
// ---------------------------------------------------------------------------
__global__ void __launch_bounds__(256) k_attn_mm(
        const float* __restrict__ x,
        const unsigned short* __restrict__ qkA,
        float* __restrict__ out) {
    __shared__ unsigned int ldsB[2][128 * 20];   // 2 x 10 KiB, u32 = 2 bf16
    const int n  = blockIdx.z;
    const int p0 = blockIdx.x * 128;
    const int d0 = blockIdx.y * 64;
    const int t    = threadIdx.x;
    const int lane = t & 31;
    const int w    = t >> 5;
    const int wm   = w >> 1;    // 0..3  (M strip)
    const int wn   = w & 1;     // 0..1  (N half)

    const float* xn = x + (size_t)n * C_ALL * HW;   // left-half channels

    v8f acc[4];
#pragma unroll
    for (int i = 0; i < 4; ++i)
#pragma unroll
        for (int r = 0; r < 8; ++r) acc[i][r] = 0.0f;

    const int mtg = (d0 >> 4) + wm;                          // global 16-row tile
    const unsigned short* aBase =
        qkA + (size_t)n * (CH * CH) + (size_t)mtg * (8 * 32 * 16);

    // -- staging: K-rows [kt*32, kt*32+32), pixels [p0, p0+128) -> bf16 LDS --
    auto stage = [&](int kt, int buf) {
        const int K0 = kt * 32;
        unsigned int* Lb = &ldsB[buf][0];
#pragma unroll
        for (int j = 0; j < 2; ++j) {
            int it = t + 256 * j;       // 0..511
            int rp = it >> 5;           // K-row pair 0..15
            int pq = it & 31;           // pixel quad 0..31 (coalesced)
            const float* src = xn + (size_t)(K0 + 2 * rp) * HW + (p0 + pq * 4);
            float4 a = *(const float4*)(src);        // row 2rp
            float4 b = *(const float4*)(src + HW);   // row 2rp+1
            // u32 slot for pixel pp, k-pair rp:  pp*20 + rp
            int base = pq * 4 * 20 + rp;
            Lb[base +  0] = pack_bf16x2(a.x, b.x);
            Lb[base + 20] = pack_bf16x2(a.y, b.y);
            Lb[base + 40] = pack_bf16x2(a.z, b.z);
            Lb[base + 60] = pack_bf16x2(a.w, b.w);
        }
    };

    stage(0, 0);
    for (int kt = 0; kt < 8; ++kt) {
        __syncthreads();                        // buf kt&1 ready for all
        if (kt < 7) stage(kt + 1, (kt + 1) & 1);  // disjoint buffer: no barrier

        // ---- A fragment: contiguous 32B per lane (pre-swizzled in ws) ----
        const unsigned short* ap = aBase + ((size_t)kt * 32 + lane) * 16;
        v8bf alo = *(const v8bf*)(ap);
        v8bf ahi = *(const v8bf*)(ap + 8);
        v16bf afrag = __builtin_shufflevector(alo, ahi,
            0,1,2,3,4,5,6,7,8,9,10,11,12,13,14,15);

        const unsigned short* Lh = (const unsigned short*)&ldsB[kt & 1][0];
#pragma unroll
        for (int nt = 0; nt < 4; ++nt) {
            // B fragment: N = lane&15, K = h + 16*(lane>>4)
            int prow = wn * 64 + nt * 16 + (lane & 15);
            const unsigned short* bp = &Lh[prow * 40 + ((lane >> 4) << 4)];
            v8bf blo = *(const v8bf*)(bp);
            v8bf bhi = *(const v8bf*)(bp + 8);
            v16bf bfrag = __builtin_shufflevector(blo, bhi,
                0,1,2,3,4,5,6,7,8,9,10,11,12,13,14,15);
            acc[nt] = __builtin_amdgcn_wmma_f32_16x16x32_bf16(
                false, afrag, false, bfrag, (short)0, acc[nt], false, false);
        }
    }

    // ---- epilogue: out = xl * (1 + att) ----
#pragma unroll
    for (int nt = 0; nt < 4; ++nt) {
        int p = p0 + wn * 64 + nt * 16 + (lane & 15);
#pragma unroll
        for (int r = 0; r < 8; ++r) {
            int d = d0 + wm * 16 + r + ((lane >> 4) << 3);   // C/D layout
            size_t o = (size_t)n * C_ALL * HW + (size_t)d * HW + p;
            out[o] = x[o] * (1.0f + acc[nt][r]);
        }
    }
}

// ---------------------------------------------------------------------------
// Kernel 4: right half, out = xr * (al + 2).  One block per (n,c).
// ---------------------------------------------------------------------------
__global__ void __launch_bounds__(256) k_right(const float* __restrict__ x,
                                               const float* __restrict__ scale_r,
                                               float* __restrict__ out) {
    const int b = blockIdx.x;         // n*256 + c
    const int n = b >> 8;
    const int c = b & 255;
    const float s = scale_r[b];
    const size_t base = ((size_t)n * C_ALL + CH + c) * HW;
    const float4* xv = (const float4*)(x + base);
    float4* ov = (float4*)(out + base);
    const int t = threadIdx.x;
#pragma unroll
    for (int j = 0; j < 4; ++j) {
        float4 v = xv[t + 256 * j];
        float4 r; r.x = v.x * s; r.y = v.y * s; r.z = v.z * s; r.w = v.w * s;
        ov[t + 256 * j] = r;
    }
}

// ---------------------------------------------------------------------------
extern "C" void kernel_launch(void* const* d_in, const int* in_sizes, int n_in,
                              void* d_out, int out_size, void* d_ws, size_t ws_size,
                              hipStream_t stream) {
    const float* x     = (const float*)d_in[0];
    const float* w_q   = (const float*)d_in[1];
    const float* b_q   = (const float*)d_in[2];
    const float* w_loc = (const float*)d_in[3];
    const float* b_loc = (const float*)d_in[4];
    float* out = (float*)d_out;

    // workspace layout (all offsets >= 64B aligned):
    float* gbuf  = (float*)d_ws;                       // 32*512 f32  = 64 KiB
    float* srbuf = gbuf + N_B * C_ALL;                 // 32*256 f32  = 32 KiB
    unsigned short* qkA = (unsigned short*)(srbuf + N_B * CH);  // 4 MiB bf16

    k_mean<<<N_B * C_ALL, 256, 0, stream>>>(x, gbuf);
    k_attn_prep<<<N_B, 256, 0, stream>>>(gbuf, w_q, b_q, w_loc, b_loc, srbuf, qkA);
    dim3 g3(HW / 128, CH / 64, N_B);
    k_attn_mm<<<g3, 256, 0, stream>>>(x, qkA, out);
    k_right<<<N_B * CH, 256, 0, stream>>>(x, srbuf, out);
}
